// YuaDecoderLayer_61881888800984
// MI455X (gfx1250) — matrix operand order repaired and checked by reading, not compiled, not run on hardware
//
#include <hip/hip_runtime.h>
#include <hip/hip_bf16.h>

// Problem constants (match reference)
#define B_   2
#define S_   2048
#define H_   2048
#define NH_  16
#define NKV_ 4
#define HD_  128
#define E_   8
#define I_   1024
#define T_   (B_ * S_)

typedef __bf16 bf16x16 __attribute__((ext_vector_type(16)));
typedef __bf16 bf16x2  __attribute__((ext_vector_type(2)));
typedef float  f32x8   __attribute__((ext_vector_type(8)));

union Frag {
  bf16x16 v;
  unsigned int u[8];
};

__device__ __forceinline__ unsigned short f2bf(float x) {
  unsigned int u = __float_as_uint(x);
  u += 0x7FFFu + ((u >> 16) & 1u);          // round-to-nearest-even
  return (unsigned short)(u >> 16);
}
// Packed 2xFP32 -> 2xBF16 (one v_cvt_pk_bf16_f32 when available)
__device__ __forceinline__ unsigned int pack_bf2(float lo, float hi) {
#if __has_builtin(__builtin_amdgcn_cvt_pk_bf16_f32)
  union { bf16x2 h; unsigned int u; } r;
  r.h = __builtin_amdgcn_cvt_pk_bf16_f32(lo, hi);
  return r.u;
#else
  return (unsigned int)f2bf(lo) | ((unsigned int)f2bf(hi) << 16);
#endif
}
__device__ __forceinline__ float bf2f(unsigned short h) {
  return __uint_as_float(((unsigned int)h) << 16);
}
__device__ __forceinline__ f32x8 fzero8() {
  f32x8 z;
#pragma unroll
  for (int i = 0; i < 8; ++i) z[i] = 0.0f;
  return z;
}

// CDNA5 async memory->LDS copy of 16 bytes (per lane). Tracked by ASYNCcnt.
__device__ __forceinline__ void async_copy_b128(const void* gptr, void* lptr) {
  unsigned long long ga = (unsigned long long)(size_t)gptr;
  unsigned int la = (unsigned int)(unsigned long long)lptr;   // low 32b = LDS offset
  asm volatile("global_load_async_to_lds_b128 %0, %1, off"
               :: "v"(la), "v"(ga) : "memory");
}
__device__ __forceinline__ void wait_async0() {
#if __has_builtin(__builtin_amdgcn_s_wait_asynccnt)
  __builtin_amdgcn_s_wait_asynccnt(0);
#else
  asm volatile("s_wait_asynccnt 0x0" ::: "memory");
#endif
}

// A fragment (16x32 bf16, MxK), contiguous K order, same as B: lane L holds row M=L%16.
// lanes 0-15 : K = {0..7} in v0..v3 (2/VGPR), {8..15} in v4..v7
// lanes 16-31: K = {16..23} in v0..v3,         {24..31} in v4..v7
__device__ __forceinline__ void load_afrag(const unsigned short* rowp, Frag& f) {
  const unsigned int lane = threadIdx.x & 31u;
  const unsigned int* p = (const unsigned int*)rowp;  // bf16 pairs
  const int bp = (lane >> 4) ? 8 : 0;
#pragma unroll
  for (int j = 0; j < 4; ++j) f.u[j] = p[bp + j];
#pragma unroll
  for (int j = 0; j < 4; ++j) f.u[4 + j] = p[bp + 4 + j];
}

// B fragment (32x16 bf16, KxN): lane L holds column N=L%16.
// lanes 0-15 hold K=0..15 sequentially, lanes 16-31 hold K=16..31
// (SWMMAC 64x16 B layout halved). rowp -> Bt row (one B column, 32 K values).
__device__ __forceinline__ void load_bfrag(const unsigned short* rowp, Frag& f) {
  const unsigned int lane = threadIdx.x & 31u;
  const unsigned int* p = (const unsigned int*)rowp;
  const int bp = (lane >> 4) ? 8 : 0;
#pragma unroll
  for (int j = 0; j < 8; ++j) f.u[j] = p[bp + j];
}

__device__ __forceinline__ f32x8 wmma_bf16(const Frag& a, const Frag& b, f32x8 c) {
  return __builtin_amdgcn_wmma_f32_16x16x32_bf16(false, a.v, false, b.v,
                                                 (short)0, c, false, false);
}

// ---------------------------------------------------------------------------
// Tiled GEMM: C[M,N] (fp32) = A[M,K] (bf16) x B[K,N] (fp32 -> bf16 on stage)
// 128x128 block tile, 256 threads = 8 waves (4x2), each wave 32x64 via
// 2x4 v_wmma_f32_16x16x32_bf16 per K-step. A tile staged with async B128
// direct-to-LDS; B tile via coalesced fp32 loads + v_cvt_pk_bf16_f32 packed stores.
// ---------------------------------------------------------------------------
__global__ __launch_bounds__(256) void gemm_bf16_kernel(
    const unsigned short* __restrict__ A,
    const float* __restrict__ Bm,
    float* __restrict__ C,
    const float* __restrict__ addsrc,
    const float* __restrict__ rowscale,
    int M, int N, int K, int rs_stride, int accum)
{
  __shared__ unsigned short As[128][40];   // [m][k], row = 80B (16B aligned)
  __shared__ unsigned short Bt[128][40];   // [n][k]

  const int tid  = threadIdx.x;
  const int lane = tid & 31;
  const int wave = tid >> 5;
  const int half = lane >> 4;
  const int l15  = lane & 15;
  const int wm = wave & 3;    // 32-row slab
  const int wn = wave >> 2;   // 64-col slab
  const int br = blockIdx.y * 128;
  const int bc = blockIdx.x * 128;

  f32x8 acc[2][4];
#pragma unroll
  for (int mt = 0; mt < 2; ++mt)
#pragma unroll
    for (int nt = 0; nt < 4; ++nt) acc[mt][nt] = fzero8();

  for (int kk = 0; kk < K; kk += 32) {
    // A tile 128x32 bf16: async direct-to-LDS, 16B per issue
#pragma unroll
    for (int i = 0; i < 2; ++i) {
      int s = i * 256 + tid;                  // 512 slots of 8 elems
      int r = s >> 2, c8 = (s & 3) * 8;
      async_copy_b128(&A[(size_t)(br + r) * K + kk + c8], &As[r][c8]);
    }
    // B tile 32x128 fp32 -> bf16 transposed: coalesced loads, packed stores
#pragma unroll
    for (int i = 0; i < 4; ++i) {
      int s = i * 256 + tid;                  // 1024 slots of 4 k-values
      int n = s & 127, k4 = (s >> 7) * 4;
      const float* bp = &Bm[(size_t)(kk + k4) * N + bc + n];
      float f0 = bp[0];
      float f1 = bp[(size_t)N];
      float f2 = bp[(size_t)2 * N];
      float f3 = bp[(size_t)3 * N];
      uint2 d;
      d.x = pack_bf2(f0, f1);
      d.y = pack_bf2(f2, f3);
      *(uint2*)&Bt[n][k4] = d;
    }
    if (kk + 32 < K)                          // prefetch next weight tile
      __builtin_prefetch(&Bm[(size_t)(kk + 32) * N + bc + (tid & 127)], 0, 1);
    wait_async0();
    __syncthreads();

    Frag a[2], b[4];
#pragma unroll
    for (int mt = 0; mt < 2; ++mt) load_afrag(&As[wm * 32 + mt * 16 + l15][0], a[mt]);
#pragma unroll
    for (int nt = 0; nt < 4; ++nt) load_bfrag(&Bt[wn * 64 + nt * 16 + l15][0], b[nt]);
#pragma unroll
    for (int mt = 0; mt < 2; ++mt)
#pragma unroll
      for (int nt = 0; nt < 4; ++nt)
        acc[mt][nt] = wmma_bf16(a[mt], b[nt], acc[mt][nt]);
    __syncthreads();
  }

  // Epilogue. C/D layout: VGPR v -> row v + 8*half; lane&15 -> col.
#pragma unroll
  for (int mt = 0; mt < 2; ++mt)
#pragma unroll
    for (int nt = 0; nt < 4; ++nt)
#pragma unroll
      for (int v = 0; v < 8; ++v) {
        int r = br + wm * 32 + mt * 16 + half * 8 + v;
        int c = bc + wn * 64 + nt * 16 + l15;
        float val = acc[mt][nt][v];
        if (addsrc)   val += addsrc[(size_t)r * N + c];
        if (rowscale) val *= rowscale[(size_t)r * rs_stride];
        size_t o = (size_t)r * N + c;
        if (accum) C[o] += val; else C[o] = val;
      }
}

// ---------------------------------------------------------------------------
// RMSNorm: one block per token row, fp32 in -> bf16 out (vectorized)
// ---------------------------------------------------------------------------
__global__ __launch_bounds__(256) void rmsnorm_kernel(
    const float* __restrict__ x, const float* __restrict__ w,
    unsigned short* __restrict__ out)
{
  const int row = blockIdx.x;
  const float4* xr4 = (const float4*)(x + (size_t)row * H_);
  float s = 0.f;
#pragma unroll
  for (int i = threadIdx.x; i < H_ / 4; i += 256) {
    float4 v = xr4[i];
    s += v.x * v.x + v.y * v.y + v.z * v.z + v.w * v.w;
  }
#pragma unroll
  for (int off = 16; off > 0; off >>= 1) s += __shfl_down(s, off, 32);
  __shared__ float red[8];
  if ((threadIdx.x & 31) == 0) red[threadIdx.x >> 5] = s;
  __syncthreads();
  if (threadIdx.x == 0) {
    float t = 0.f;
#pragma unroll
    for (int i = 0; i < 8; ++i) t += red[i];
    red[0] = rsqrtf(t / (float)H_ + 1e-5f);
  }
  __syncthreads();
  float rinv = red[0];
  const float4* wr4 = (const float4*)w;
  uint2* o4 = (uint2*)(out + (size_t)row * H_);
  for (int i = threadIdx.x; i < H_ / 4; i += 256) {
    float4 v = xr4[i];
    float4 ww = wr4[i];
    uint2 d;
    d.x = pack_bf2(ww.x * v.x * rinv, ww.y * v.y * rinv);
    d.y = pack_bf2(ww.z * v.z * rinv, ww.w * v.w * rinv);
    o4[i] = d;
  }
}

// ---------------------------------------------------------------------------
// RoPE + layout transform: [B,S,nh,HD] fp32 -> [B,nh,S,HD] bf16
// 32 threads: thread t handles dims {2t,2t+1} and {2t+64,2t+65} (packed stores)
// ---------------------------------------------------------------------------
__global__ __launch_bounds__(32) void rope_convert_kernel(
    const float* __restrict__ src, unsigned short* __restrict__ dst,
    int nh, int do_rope)
{
  const int s = blockIdx.x, h = blockIdx.y, b = blockIdx.z;
  const int d2 = threadIdx.x * 2;          // 0..62, even
  const float* in = src + (((size_t)b * S_ + s) * nh + h) * HD_;
  unsigned short* out = dst + (((size_t)b * nh + h) * S_ + s) * HD_;
  float2 x1 = *(const float2*)&in[d2];
  float2 x2 = *(const float2*)&in[d2 + 64];
  if (do_rope) {
    // inv_freq = theta^(-d/64);  ln(500000) = 13.122363377404328
    float i0 = __expf(-(float)(d2    ) * (13.122363377404328f / 64.0f));
    float i1 = __expf(-(float)(d2 + 1) * (13.122363377404328f / 64.0f));
    float f0 = (float)s * i0, f1 = (float)s * i1;
    float c0 = cosf(f0), s0 = sinf(f0);
    float c1 = cosf(f1), s1 = sinf(f1);
    *(unsigned int*)&out[d2]      = pack_bf2(x1.x * c0 - x2.x * s0,
                                             x1.y * c1 - x2.y * s1);
    *(unsigned int*)&out[d2 + 64] = pack_bf2(x2.x * c0 + x1.x * s0,
                                             x2.y * c1 + x1.y * s1);
  } else {
    *(unsigned int*)&out[d2]      = pack_bf2(x1.x, x1.y);
    *(unsigned int*)&out[d2 + 64] = pack_bf2(x2.x, x2.y);
  }
}

// ---------------------------------------------------------------------------
// Flash attention (causal, GQA): 64-query tile per block, stream 32-key blocks.
// Scores and PV via v_wmma_f32_16x16x32_bf16; Q/K tiles staged with async
// direct-to-LDS B128 copies; online softmax in LDS.
// ---------------------------------------------------------------------------
__global__ __launch_bounds__(256) void attn_kernel(
    const unsigned short* __restrict__ Qg,   // [B*NH, S, HD] bf16
    const unsigned short* __restrict__ Kg,   // [B*NKV, S, HD] bf16 (RoPE'd)
    const unsigned short* __restrict__ Vg,   // [B*NKV, S, HD] bf16
    unsigned short* __restrict__ Og)         // [B*S, NH*HD] bf16
{
  const int qblk = blockIdx.x;
  const int h    = blockIdx.y;
  const int b    = blockIdx.z;
  const int hk   = h / (NH_ / NKV_);
  const int tid  = threadIdx.x;
  const int lane = tid & 31;
  const int wave = tid >> 5;
  const int half = lane >> 4;
  const int l15  = lane & 15;
  const int wmt  = wave & 3;    // 16-row tile
  const int wct  = wave >> 2;   // col group (scores: 16 cols; O: 64 cols)

  __shared__ unsigned short Qs[64][136];   // 272B rows (16B aligned)
  __shared__ unsigned short Ks[32][136];
  __shared__ unsigned short Vt[128][40];   // [d][key], 80B rows
  __shared__ float          Sl[64][36];
  __shared__ unsigned short Pl[64][34];
  __shared__ float          alpha_l[64];
  __shared__ float          linv_l[64];

  const unsigned short* qbase =
      Qg + (((size_t)b * NH_ + h) * S_ + (size_t)qblk * 64) * HD_;
#pragma unroll
  for (int i = 0; i < 4; ++i) {            // Q tile 64x128: async to LDS
    int s = i * 256 + tid;                 // 1024 slots of 8
    int r = s >> 4, c8 = (s & 15) * 8;
    async_copy_b128(&qbase[r * HD_ + c8], &Qs[r][c8]);
  }

  f32x8 oacc[4];
#pragma unroll
  for (int nt = 0; nt < 4; ++nt) oacc[nt] = fzero8();
  float m_i = -1e30f, l_i = 0.f;           // valid for tid < 64 (row owner)

  const float scale = 0.08838834764831845f;  // 1/sqrt(HD)
  const int nkb = 2 * qblk + 2;              // 32-key blocks under causal mask

  for (int kb = 0; kb < nkb; ++kb) {
    const unsigned short* kbase =
        Kg + (((size_t)b * NKV_ + hk) * S_ + (size_t)kb * 32) * HD_;
    const unsigned short* vbase =
        Vg + (((size_t)b * NKV_ + hk) * S_ + (size_t)kb * 32) * HD_;
#pragma unroll
    for (int i = 0; i < 2; ++i) {          // K tile 32x128: async to LDS
      int s = i * 256 + tid;
      int r = s >> 4, c8 = (s & 15) * 8;
      async_copy_b128(&kbase[r * HD_ + c8], &Ks[r][c8]);
    }
#pragma unroll
    for (int i = 0; i < 2; ++i) {          // V tile: wide load + transpose
      int s = i * 256 + tid;
      int r = s >> 4, c8 = (s & 15) * 8;   // r = key, c8 = d base
      uint4 dv = *(const uint4*)&vbase[r * HD_ + c8];
      const unsigned short* vv = (const unsigned short*)&dv;
#pragma unroll
      for (int j = 0; j < 8; ++j) Vt[c8 + j][r] = vv[j];
    }
    wait_async0();
    __syncthreads();

    // Scores: wave owns rows [wmt*16,+16), cols [wct*16,+16); K = HD = 128
    f32x8 sacc = fzero8();
#pragma unroll
    for (int kk = 0; kk < HD_; kk += 32) {
      Frag a;  load_afrag(&Qs[wmt * 16 + l15][kk], a);
      Frag bb; load_bfrag(&Ks[wct * 16 + l15][kk], bb);
      sacc = wmma_bf16(a, bb, sacc);
    }
#pragma unroll
    for (int v = 0; v < 8; ++v) {
      int r = wmt * 16 + half * 8 + v;
      int c = wct * 16 + l15;
      float val = sacc[v] * scale;
      if (kb * 32 + c > qblk * 64 + r) val = -1e30f;   // causal mask
      Sl[r][c] = val;
    }
    __syncthreads();

    // Online softmax: thread r (<64) owns row r; m/l persist in registers.
    if (tid < 64) {
      float mn = m_i;
      for (int c = 0; c < 32; ++c) mn = fmaxf(mn, Sl[tid][c]);
      float al = expf(m_i - mn);
      float ssum = 0.f;
      for (int c = 0; c < 32; c += 2) {
        float p0 = expf(Sl[tid][c]     - mn);
        float p1 = expf(Sl[tid][c + 1] - mn);
        ssum += p0 + p1;
        *(unsigned int*)&Pl[tid][c] = pack_bf2(p0, p1);
      }
      m_i = mn;
      l_i = l_i * al + ssum;
      alpha_l[tid] = al;
    }
    __syncthreads();

    // Rescale O accumulators, then O += P x V (K = 32 keys)
#pragma unroll
    for (int nt = 0; nt < 4; ++nt)
#pragma unroll
      for (int v = 0; v < 8; ++v)
        oacc[nt][v] *= alpha_l[wmt * 16 + half * 8 + v];

    {
      Frag a; load_afrag(&Pl[wmt * 16 + l15][0], a);
#pragma unroll
      for (int nt = 0; nt < 4; ++nt) {
        Frag bb; load_bfrag(&Vt[wct * 64 + nt * 16 + l15][0], bb);
        oacc[nt] = wmma_bf16(a, bb, oacc[nt]);
      }
    }
    __syncthreads();
  }

  if (tid < 64) linv_l[tid] = 1.0f / l_i;
  __syncthreads();

#pragma unroll
  for (int nt = 0; nt < 4; ++nt)
#pragma unroll
    for (int v = 0; v < 8; ++v) {
      int r = wmt * 16 + half * 8 + v;
      int c = wct * 64 + nt * 16 + l15;
      float val = oacc[nt][v] * linv_l[r];
      size_t row = (size_t)b * S_ + (size_t)qblk * 64 + r;
      Og[row * (NH_ * HD_) + (size_t)h * HD_ + c] = f2bf(val);
    }
}

// ---------------------------------------------------------------------------
// Router: logits = x @ router_w, relu, top-2, normalized combine weights [T,E]
// ---------------------------------------------------------------------------
__global__ __launch_bounds__(256) void router_kernel(
    const unsigned short* __restrict__ xn, const float* __restrict__ rw,
    float* __restrict__ combine)
{
  int t = blockIdx.x * 256 + threadIdx.x;
  if (t >= T_) return;
  float acc[E_];
#pragma unroll
  for (int e = 0; e < E_; ++e) acc[e] = 0.f;
  const unsigned short* xr = xn + (size_t)t * H_;
  for (int hh = 0; hh < H_; ++hh) {
    float xv = bf2f(xr[hh]);
    const float* r = rw + (size_t)hh * E_;
#pragma unroll
    for (int e = 0; e < E_; ++e) acc[e] += xv * r[e];
  }
#pragma unroll
  for (int e = 0; e < E_; ++e) acc[e] = fmaxf(acc[e], 0.f);
  int i0 = 0; float v0 = acc[0];
  for (int e = 1; e < E_; ++e) if (acc[e] > v0) { v0 = acc[e]; i0 = e; }
  int i1 = -1; float v1 = -1.f;
  for (int e = 0; e < E_; ++e) if (e != i0 && acc[e] > v1) { v1 = acc[e]; i1 = e; }
  float denom = v0 + v1 + 1e-6f;
#pragma unroll
  for (int e = 0; e < E_; ++e) combine[(size_t)t * E_ + e] = 0.f;
  combine[(size_t)t * E_ + i0] = v0 / denom;
  combine[(size_t)t * E_ + i1] += v1 / denom;
}

// 2 elements per thread, packed bf16 store
__global__ __launch_bounds__(256) void silu_mul_kernel(
    const float* __restrict__ g, const float* __restrict__ u,
    unsigned short* __restrict__ out, int n)
{
  int i2 = (blockIdx.x * 256 + threadIdx.x) * 2;
  if (i2 >= n) return;
  float g0 = g[i2], g1 = g[i2 + 1];
  float r0 = (g0 / (1.0f + __expf(-g0))) * u[i2];
  float r1 = (g1 / (1.0f + __expf(-g1))) * u[i2 + 1];
  *(unsigned int*)&out[i2] = pack_bf2(r0, r1);
}

// float4 residual add
__global__ __launch_bounds__(256) void add2_kernel(
    const float* __restrict__ a, const float* __restrict__ b,
    float* __restrict__ o, int n)
{
  int i4 = (blockIdx.x * 256 + threadIdx.x) * 4;
  if (i4 >= n) return;
  float4 va = *(const float4*)&a[i4];
  float4 vb = *(const float4*)&b[i4];
  float4 vo;
  vo.x = va.x + vb.x; vo.y = va.y + vb.y;
  vo.z = va.z + vb.z; vo.w = va.w + vb.w;
  *(float4*)&o[i4] = vo;
}

// ---------------------------------------------------------------------------
extern "C" void kernel_launch(void* const* d_in, const int* in_sizes, int n_in,
                              void* d_out, int out_size, void* d_ws, size_t ws_size,
                              hipStream_t stream) {
  const float* hs  = (const float*)d_in[0];
  const float* ln1 = (const float*)d_in[1];
  const float* ln2 = (const float*)d_in[2];
  const float* wq  = (const float*)d_in[3];
  const float* wk  = (const float*)d_in[4];
  const float* wv  = (const float*)d_in[5];
  const float* wo  = (const float*)d_in[6];
  const float* rw  = (const float*)d_in[7];
  const float* eg  = (const float*)d_in[8];
  const float* eu  = (const float*)d_in[9];
  const float* ed  = (const float*)d_in[10];
  const float* sg  = (const float*)d_in[11];
  const float* su  = (const float*)d_in[12];
  const float* sd  = (const float*)d_in[13];
  float* out = (float*)d_out;

  char* p = (char*)d_ws;
  auto alloc = [&](size_t bytes) -> void* {
    void* r = (void*)p;
    p += (bytes + 255) & ~(size_t)255;
    return r;
  };
  unsigned short* xn1   = (unsigned short*)alloc((size_t)T_ * H_ * 2);
  unsigned short* xn2   = (unsigned short*)alloc((size_t)T_ * H_ * 2);
  float* qf             = (float*)alloc((size_t)T_ * NH_ * HD_ * 4);
  float* kf             = (float*)alloc((size_t)T_ * NKV_ * HD_ * 4);
  float* vf             = (float*)alloc((size_t)T_ * NKV_ * HD_ * 4);
  unsigned short* qb    = (unsigned short*)alloc((size_t)T_ * NH_ * HD_ * 2);
  unsigned short* kb    = (unsigned short*)alloc((size_t)T_ * NKV_ * HD_ * 2);
  unsigned short* vb    = (unsigned short*)alloc((size_t)T_ * NKV_ * HD_ * 2);
  unsigned short* attnb = (unsigned short*)alloc((size_t)T_ * NH_ * HD_ * 2);
  float* hiddenf        = (float*)alloc((size_t)T_ * H_ * 4);
  float* gf             = (float*)alloc((size_t)T_ * I_ * 4);
  float* uf             = (float*)alloc((size_t)T_ * I_ * 4);
  unsigned short* hb    = (unsigned short*)alloc((size_t)T_ * I_ * 2);
  float* moef           = (float*)alloc((size_t)T_ * H_ * 4);
  float* comb           = (float*)alloc((size_t)T_ * E_ * 4);

  // 1) ln1 -> bf16
  rmsnorm_kernel<<<T_, 256, 0, stream>>>(hs, ln1, xn1);
  // 2) QKV projections (WMMA)
  gemm_bf16_kernel<<<dim3((NH_*HD_)/128, T_/128), 256, 0, stream>>>(
      xn1, wq, qf, nullptr, nullptr, T_, NH_*HD_, H_, 0, 0);
  gemm_bf16_kernel<<<dim3((NKV_*HD_)/128, T_/128), 256, 0, stream>>>(
      xn1, wk, kf, nullptr, nullptr, T_, NKV_*HD_, H_, 0, 0);
  gemm_bf16_kernel<<<dim3((NKV_*HD_)/128, T_/128), 256, 0, stream>>>(
      xn1, wv, vf, nullptr, nullptr, T_, NKV_*HD_, H_, 0, 0);
  // 3) RoPE + [B,h,S,D] layout, bf16
  rope_convert_kernel<<<dim3(S_, NH_,  B_), 32, 0, stream>>>(qf, qb, NH_, 1);
  rope_convert_kernel<<<dim3(S_, NKV_, B_), 32, 0, stream>>>(kf, kb, NKV_, 1);
  rope_convert_kernel<<<dim3(S_, NKV_, B_), 32, 0, stream>>>(vf, vb, NKV_, 0);
  // 4) Causal GQA flash attention (WMMA + async-to-LDS)
  attn_kernel<<<dim3(S_/64, NH_, B_), 256, 0, stream>>>(qb, kb, vb, attnb);
  // 5) O-projection + residual -> hidden
  gemm_bf16_kernel<<<dim3(H_/128, T_/128), 256, 0, stream>>>(
      attnb, wo, hiddenf, hs, nullptr, T_, H_, NH_*HD_, 0, 0);
  // 6) ln2 -> bf16
  rmsnorm_kernel<<<T_, 256, 0, stream>>>(hiddenf, ln2, xn2);
  // 7) router combine weights
  router_kernel<<<T_/256, 256, 0, stream>>>(xn2, rw, comb);
  // 8) shared expert: silu(x@sg)*(x@su) @ sd  -> moef (store)
  gemm_bf16_kernel<<<dim3(I_/128, T_/128), 256, 0, stream>>>(
      xn2, sg, gf, nullptr, nullptr, T_, I_, H_, 0, 0);
  gemm_bf16_kernel<<<dim3(I_/128, T_/128), 256, 0, stream>>>(
      xn2, su, uf, nullptr, nullptr, T_, I_, H_, 0, 0);
  silu_mul_kernel<<<(T_*I_)/512, 256, 0, stream>>>(gf, uf, hb, T_*I_);
  gemm_bf16_kernel<<<dim3(H_/128, T_/128), 256, 0, stream>>>(
      hb, sd, moef, nullptr, nullptr, T_, H_, I_, 0, 0);
  // 9) routed experts, dense like the reference; combine-weighted accumulate.
  //    Sequential launches on one stream make moef += race-free.
  for (int e = 0; e < E_; ++e) {
    const float* ege = eg + (size_t)e * H_ * I_;
    const float* eue = eu + (size_t)e * H_ * I_;
    const float* ede = ed + (size_t)e * I_ * H_;
    gemm_bf16_kernel<<<dim3(I_/128, T_/128), 256, 0, stream>>>(
        xn2, ege, gf, nullptr, nullptr, T_, I_, H_, 0, 0);
    gemm_bf16_kernel<<<dim3(I_/128, T_/128), 256, 0, stream>>>(
        xn2, eue, uf, nullptr, nullptr, T_, I_, H_, 0, 0);
    silu_mul_kernel<<<(T_*I_)/512, 256, 0, stream>>>(gf, uf, hb, T_*I_);
    gemm_bf16_kernel<<<dim3(H_/128, T_/128), 256, 0, stream>>>(
        hb, ede, moef, nullptr, comb + e, T_, H_, I_, E_, 1);
  }
  // 10) out = hidden + (shared + routed)
  add2_kernel<<<(T_*H_)/1024, 256, 0, stream>>>(hiddenf, moef, out, T_*H_);

  (void)in_sizes; (void)n_in; (void)out_size; (void)ws_size;
}
